// GVMLayer_5093831213706
// MI455X (gfx1250) — matrix operand order repaired and checked by reading, not compiled, not run on hardware
//
#include <hip/hip_runtime.h>
#include <cstddef>

// ---------------- problem constants ----------------
#define ND    65536      // total nodes
#define DH    128        // hidden dim
#define EE    (ND * 8)   // edges
#define BB    64         // graphs
#define NPER  1024       // nodes per graph
#define KS    32         // PMA seeds

typedef __bf16 bf16;
typedef __attribute__((ext_vector_type(16))) __bf16 v16bf;
typedef __attribute__((ext_vector_type(8)))  float  v8f;

__device__ inline v8f zero8() {
    v8f c;
#pragma unroll
    for (int i = 0; i < 8; ++i) c[i] = 0.0f;
    return c;
}

__device__ inline v8f wmma_bf(v16bf a, v16bf b, v8f c) {
    // D = A(16x32 bf16) * B(32x16 bf16) + C(16x16 f32)
    return __builtin_amdgcn_wmma_f32_16x16x32_bf16(false, a, false, b, (short)0, c, false, false);
}

// A-fragment element->K permutation (16-bit A 16x32 layout):
// lanes 0-15: elems 0-7 -> K 0-7, elems 8-15 -> K 16-23
// lanes16-31: elems 0-7 -> K 8-15, elems 8-15 -> K 24-31
__device__ inline int a_kl(int e, int kh) { return ((e < 8) ? e : e + 8) + kh * 8; }

// Swizzled LDS index for B operands so each lane's 16 elements are contiguous.
// B-frag: lane = kh*16+l16 -> (col = nt*16+l16, Krow = ks*32 + kh*8 + (e<8 ? e : e+8))
__device__ inline int bswz(int kk, int nn, int nc) {
    int ks = kk >> 5, kl = kk & 31;
    int kh = (kl >> 3) & 1, e = (kl & 7) | ((kl >> 4) << 3);
    int nt = nn >> 4, l16 = nn & 15;
    return (((ks * (nc >> 4) + nt) * 32) + kh * 16 + l16) * 16 + e;
}

// Contiguous 16-element bf16 fragment (32B) -> two 16B loads
__device__ inline v16bf ld_frag16(const bf16* p) {
    union { uint4 q[2]; v16bf v; } u;
    u.q[0] = *(const uint4*)(p);
    u.q[1] = *(const uint4*)(p + 8);
    return u.v;
}

// A-fragment: two 8-element contiguous runs separated by 16 elements.
// p points at run0 (elements a_kl(0..7,kh)); run1 at p+16.
__device__ inline v16bf ld_frag_a(const bf16* p) {
    union { uint4 q[2]; v16bf v; } u;
    u.q[0] = *(const uint4*)(p);
    u.q[1] = *(const uint4*)(p + 16);
    return u.v;
}

// =====================================================================
// Edge kernel: msg = relu(h[src] + edge_attr); atomic scatter-add to aggr[dst]
// one wave per edge, float4 per lane (32 lanes * 4 = 128 dims)
// =====================================================================
__global__ __launch_bounds__(256)
void edge_kernel(const float* __restrict__ h, const float* __restrict__ eattr,
                 const int* __restrict__ src, const int* __restrict__ dst,
                 float* __restrict__ aggr) {
    int wid  = (blockIdx.x * 256 + threadIdx.x) >> 5;
    int lane = threadIdx.x & 31;
    if (wid >= EE) return;
    int s = src[wid], d = dst[wid];
    const float4* hs = (const float4*)(h + (size_t)s * DH);
    const float4* ea = (const float4*)(eattr + (size_t)wid * DH);
    float4 a = hs[lane], b = ea[lane];
    float4 m;
    m.x = fmaxf(a.x + b.x, 0.0f);
    m.y = fmaxf(a.y + b.y, 0.0f);
    m.z = fmaxf(a.z + b.z, 0.0f);
    m.w = fmaxf(a.w + b.w, 0.0f);
    float* ap = aggr + (size_t)d * DH + lane * 4;
    unsafeAtomicAdd(ap + 0, m.x);
    unsafeAtomicAdd(ap + 1, m.y);
    unsafeAtomicAdd(ap + 2, m.z);
    unsafeAtomicAdd(ap + 3, m.w);
}

// =====================================================================
// Fused GINE MLP: Z = relu(((1+eps)*h + aggr) @ W1 + b1) @ W2 + b2
// Also accumulates per-channel sum/sumsq for batchnorm.
// Block = 64 rows; 8 waves; WMMA bf16; weights pre-swizzled in LDS.
// =====================================================================
__global__ __launch_bounds__(256)
void gine_mlp_kernel(const float* __restrict__ h, const float* __restrict__ aggr,
                     const float* __restrict__ epsp,
                     const float* __restrict__ W1, const float* __restrict__ b1,
                     const float* __restrict__ W2, const float* __restrict__ b2,
                     float* __restrict__ Z, float* __restrict__ bnsums) {
    extern __shared__ char smem_raw[];
    bf16* sA   = (bf16*)smem_raw;        // 64*128  (row-major)
    bf16* sW1  = sA  + 64 * 128;         // 128*128 (B-swizzled)
    bf16* sW2  = sW1 + 128 * 128;        // 128*128 (B-swizzled)
    bf16* sMid = sW2 + 128 * 128;        // 64*128  (row-major)
    float* sSum = (float*)(sMid + 64 * 128); // 256 floats

    const int tid = threadIdx.x;
    const float alpha = 1.0f + epsp[0];
    const size_t row0 = (size_t)blockIdx.x * 64;

    for (int i = tid; i < 128 * 128; i += 256) {
        int kk = i >> 7, nn = i & 127;
        int di = bswz(kk, nn, 128);
        sW1[di] = (bf16)W1[i];
        sW2[di] = (bf16)W2[i];
    }
    for (int i = tid; i < 64 * 128; i += 256) {
        size_t g = row0 * DH + (size_t)i;
        sA[i] = (bf16)(alpha * h[g] + aggr[g]);
    }
    sSum[tid] = 0.0f;
    __syncthreads();

    const int wave = tid >> 5, lane = tid & 31, kh = lane >> 4, l16 = lane & 15;
    const int mtile = wave >> 1, nc0 = (wave & 1) * 64;
    const int nt0 = nc0 >> 4;
    const int arow = mtile * 16 + l16;

    v8f acc[4];
#pragma unroll
    for (int t = 0; t < 4; ++t) acc[t] = zero8();
#pragma unroll
    for (int ks = 0; ks < 4; ++ks) {
        v16bf af = ld_frag_a(&sA[arow * 128 + ks * 32 + kh * 8]);
#pragma unroll
        for (int t = 0; t < 4; ++t) {
            v16bf bv = ld_frag16(&sW1[((ks * 8 + nt0 + t) * 32 + lane) * 16]);
            acc[t] = wmma_bf(af, bv, acc[t]);
        }
    }
#pragma unroll
    for (int t = 0; t < 4; ++t) {
        int col = nc0 + t * 16 + l16;
        float bias = b1[col];
#pragma unroll
        for (int r = 0; r < 8; ++r) {
            int mr = mtile * 16 + kh * 8 + r;
            sMid[mr * 128 + col] = (bf16)fmaxf(acc[t][r] + bias, 0.0f);
        }
    }
    __syncthreads();

#pragma unroll
    for (int t = 0; t < 4; ++t) acc[t] = zero8();
#pragma unroll
    for (int ks = 0; ks < 4; ++ks) {
        v16bf af = ld_frag_a(&sMid[arow * 128 + ks * 32 + kh * 8]);
#pragma unroll
        for (int t = 0; t < 4; ++t) {
            v16bf bv = ld_frag16(&sW2[((ks * 8 + nt0 + t) * 32 + lane) * 16]);
            acc[t] = wmma_bf(af, bv, acc[t]);
        }
    }
#pragma unroll
    for (int t = 0; t < 4; ++t) {
        int col = nc0 + t * 16 + l16;
        float bias = b2[col];
        float ps = 0.0f, pq = 0.0f;
#pragma unroll
        for (int r = 0; r < 8; ++r) {
            int mr = mtile * 16 + kh * 8 + r;
            float v = acc[t][r] + bias;
            Z[(row0 + mr) * DH + col] = v;
            ps += v; pq += v * v;
        }
        atomicAdd(&sSum[col], ps);
        atomicAdd(&sSum[128 + col], pq);
    }
    __syncthreads();
    unsafeAtomicAdd(&bnsums[tid], sSum[tid]);
}

// =====================================================================
// BatchNorm apply (training stats), optional +x residual into target
// =====================================================================
__global__ __launch_bounds__(256)
void bn_apply_kernel(const float* __restrict__ Z, const float* __restrict__ bnsums,
                     const float* __restrict__ gamma, const float* __restrict__ beta,
                     const float* __restrict__ addx, float* __restrict__ out) {
    size_t idx = (size_t)blockIdx.x * 256 + threadIdx.x;
    int c = (int)(idx & 127);
    float s  = bnsums[c];
    float sq = bnsums[128 + c];
    const float invN = 1.0f / (float)ND;
    float mu  = s * invN;
    float var = sq * invN - mu * mu;
    float v = (Z[idx] - mu) * rsqrtf(var + 1e-5f) * gamma[c] + beta[c];
    if (addx) v += addx[idx];
    out[idx] = v;
}

// =====================================================================
// Generic GEMM: C[M,NC] = act(A @ W(KD,NC) + bias) (+resid)
// Weights staged to LDS bf16 pre-swizzled; A staged to LDS bf16 row-major.
// Block = 64 rows, 8 waves, each wave 16 rows x NC/2 cols.
// =====================================================================
template <int KD, int NC>
__global__ __launch_bounds__(256)
void gemm_kernel(const float* __restrict__ A,
                 const float* __restrict__ W, int transW,
                 const float* __restrict__ bias, int M, int act,
                 const float* __restrict__ resid,
                 float* __restrict__ C, bf16* __restrict__ Cbf) {
    extern __shared__ char smem_raw[];
    bf16* sW = (bf16*)smem_raw;       // KD*NC  (B-swizzled)
    bf16* sA = sW + KD * NC;          // 64*KD  (row-major)
    const int tid = threadIdx.x;

    if (transW) {
        for (int i = tid; i < KD * NC; i += 256) {
            int nn = i / KD, kk = i % KD;
            sW[bswz(kk, nn, NC)] = (bf16)W[i];
        }
    } else {
        for (int i = tid; i < KD * NC; i += 256) {
            int kk = i / NC, nn = i % NC;
            sW[bswz(kk, nn, NC)] = (bf16)W[i];
        }
    }
    const size_t row0 = (size_t)blockIdx.x * 64;
    for (int i = tid; i < 64 * KD; i += 256) {
        int r = i / KD, kk = i % KD;
        size_t gr = row0 + r;
        float v = (gr < (size_t)M) ? A[gr * KD + kk] : 0.0f;
        sA[i] = (bf16)v;
    }
    __syncthreads();

    const int wave = tid >> 5, lane = tid & 31, kh = lane >> 4, l16 = lane & 15;
    const int mtile = wave >> 1, nc0 = (wave & 1) * (NC / 2);
    const int nt0 = nc0 >> 4;
    constexpr int NT = NC / 32;
    const int arow = mtile * 16 + l16;

    v8f acc[NT];
#pragma unroll
    for (int t = 0; t < NT; ++t) acc[t] = zero8();

#pragma unroll
    for (int ks = 0; ks < KD / 32; ++ks) {
        v16bf af = ld_frag_a(&sA[arow * KD + ks * 32 + kh * 8]);
#pragma unroll
        for (int t = 0; t < NT; ++t) {
            v16bf bv = ld_frag16(&sW[((ks * (NC >> 4) + nt0 + t) * 32 + lane) * 16]);
            acc[t] = wmma_bf(af, bv, acc[t]);
        }
    }
#pragma unroll
    for (int t = 0; t < NT; ++t) {
        int col = nc0 + t * 16 + l16;
        float b = bias ? bias[col] : 0.0f;
#pragma unroll
        for (int r = 0; r < 8; ++r) {
            size_t grow = row0 + mtile * 16 + kh * 8 + r;
            if (grow < (size_t)M) {
                float v = acc[t][r] + b;
                if (act == 1) v = fmaxf(v, 0.0f);
                if (resid) v += resid[grow * NC + col];
                if (C)   C[grow * NC + col] = v;
                if (Cbf) Cbf[grow * NC + col] = (bf16)v;
            }
        }
    }
}

// =====================================================================
// Row LayerNorm over D=128: one wave per row
// =====================================================================
__global__ __launch_bounds__(256)
void ln_kernel(const float* __restrict__ X, const float* __restrict__ g,
               const float* __restrict__ b, float* __restrict__ Y, int rows) {
    int row  = (blockIdx.x * 256 + threadIdx.x) >> 5;
    int lane = threadIdx.x & 31;
    if (row >= rows) return;
    const float4* xr = (const float4*)(X + (size_t)row * DH);
    float4 v = xr[lane];
    float s = v.x + v.y + v.z + v.w;
#pragma unroll
    for (int m = 16; m > 0; m >>= 1) s += __shfl_xor(s, m, 32);
    float mu = s * (1.0f / 128.0f);
    float dx = v.x - mu, dy = v.y - mu, dz = v.z - mu, dw = v.w - mu;
    float ss = dx * dx + dy * dy + dz * dz + dw * dw;
#pragma unroll
    for (int m = 16; m > 0; m >>= 1) ss += __shfl_xor(ss, m, 32);
    float r = rsqrtf(ss * (1.0f / 128.0f) + 1e-5f);
    float4 o;
    o.x = dx * r * g[lane * 4 + 0] + b[lane * 4 + 0];
    o.y = dy * r * g[lane * 4 + 1] + b[lane * 4 + 1];
    o.z = dz * r * g[lane * 4 + 2] + b[lane * 4 + 2];
    o.w = dw * r * g[lane * 4 + 3] + b[lane * 4 + 3];
    ((float4*)(Y + (size_t)row * DH))[lane] = o;
}

// =====================================================================
// PMA attention: per-graph block. For each of 4 heads:
//   S = (Qh @ Kh^T)/sqrt(D)  (WMMA), softmax over 1024 nodes,
//   accumulate sum-over-heads of probs into Wsum[b][n][k],
//   O[b,k,:] = Q[k,:] + concat_h(A_h @ V_h)  (WMMA)
// Dynamic LDS: S(128KB f32) + Abf(64KB) + VT(64KB) = 256KB
// =====================================================================
__global__ __launch_bounds__(256)
void pma_attn_kernel(const bf16* __restrict__ Qbf, const float* __restrict__ Qf,
                     const bf16* __restrict__ Kb, const bf16* __restrict__ Vb,
                     float* __restrict__ O, float* __restrict__ Wsum) {
    extern __shared__ char smem_raw[];
    float* S  = (float*)smem_raw;           // 32*1024 f32
    bf16* Abf = (bf16*)(S + 32 * 1024);     // 32*1024 bf16
    bf16* VT  = Abf + 32 * 1024;            // 32*1024 bf16 (per-head V^T)

    const int b = blockIdx.x;
    const int tid = threadIdx.x, wave = tid >> 5, lane = tid & 31;
    const int kh = lane >> 4, l16 = lane & 15;
    const bf16* Kg = Kb + (size_t)b * NPER * DH;
    const bf16* Vg = Vb + (size_t)b * NPER * DH;
    float* Wg = Wsum + (size_t)b * NPER * KS;
    const float scale = 0.088388347648318447f; // 1/sqrt(128)

    for (int head = 0; head < 4; ++head) {
        // stage V^T for this head: VT[d][n]
        for (int n = tid; n < NPER; n += 256) {
            const bf16* vrow = Vg + (size_t)n * DH + head * 32;
#pragma unroll
            for (int d2 = 0; d2 < 32; ++d2) VT[d2 * NPER + n] = vrow[d2];
        }
        // Q fragments (both seed tiles) for this head
        v16bf qf[2];
#pragma unroll
        for (int mt = 0; mt < 2; ++mt)
            qf[mt] = ld_frag_a(Qbf + (size_t)(mt * 16 + l16) * DH + head * 32 + kh * 8);
        // scores: wave handles 8 node-tiles x both seed tiles
#pragma unroll
        for (int j = 0; j < 8; ++j) {
            int nt = wave * 8 + j;
            v16bf bv = ld_frag_a(Kg + (size_t)(nt * 16 + l16) * DH + head * 32 + kh * 8);
#pragma unroll
            for (int mt = 0; mt < 2; ++mt) {
                v8f c = zero8();
                c = wmma_bf(qf[mt], bv, c);
#pragma unroll
                for (int r = 0; r < 8; ++r)
                    S[(mt * 16 + kh * 8 + r) * NPER + nt * 16 + l16] = c[r] * scale;
            }
        }
        __syncthreads();
        // softmax over nodes: 8 threads per seed row
        {
            int row = tid >> 3, sub = tid & 7;
            float mx = -1e30f;
            for (int c0 = sub; c0 < NPER; c0 += 8) mx = fmaxf(mx, S[row * NPER + c0]);
#pragma unroll
            for (int m = 1; m < 8; m <<= 1) mx = fmaxf(mx, __shfl_xor(mx, m, 32));
            float sum = 0.0f;
            for (int c0 = sub; c0 < NPER; c0 += 8) {
                float e_ = __expf(S[row * NPER + c0] - mx);
                S[row * NPER + c0] = e_;
                sum += e_;
            }
#pragma unroll
            for (int m = 1; m < 8; m <<= 1) sum += __shfl_xor(sum, m, 32);
            float inv = 1.0f / sum;
            for (int c0 = sub; c0 < NPER; c0 += 8) {
                float p = S[row * NPER + c0] * inv;
                Abf[row * NPER + c0] = (bf16)p;
                float* wp = &Wg[(size_t)c0 * KS + row];
                if (head == 0) *wp = p; else *wp += p;
            }
        }
        __syncthreads();
        // out = A @ Vh : 2 seed tiles x 2 dim tiles, K=1024
        if (wave < 4) {
            int mt = wave & 1, dt = wave >> 1;
            v8f c = zero8();
            for (int ks = 0; ks < 32; ++ks) {
                v16bf af = ld_frag_a(&Abf[(mt * 16 + l16) * NPER + ks * 32 + kh * 8]);
                v16bf bv = ld_frag_a(&VT[(dt * 16 + l16) * NPER + ks * 32 + kh * 8]);
                c = wmma_bf(af, bv, c);
            }
#pragma unroll
            for (int r = 0; r < 8; ++r) {
                int k = mt * 16 + kh * 8 + r;
                int d = head * 32 + dt * 16 + l16;
                O[(size_t)b * KS * DH + (size_t)k * DH + d] = Qf[(size_t)k * DH + d] + c[r];
            }
        }
        __syncthreads();
    }
}

// =====================================================================
// Small per-graph transformer self-attention on 32 VN tokens, 8 heads x 16
// =====================================================================
__global__ __launch_bounds__(256)
void attn_small_kernel(const float* __restrict__ QKV, float* __restrict__ SA) {
    __shared__ float sq[32 * 128];
    __shared__ float sk[32 * 128];
    __shared__ float sv[32 * 128];
    const int b = blockIdx.x, tid = threadIdx.x;
    const float* base = QKV + (size_t)b * KS * 384;
    for (int i = tid; i < 32 * 128; i += 256) {
        int tok = i >> 7, d = i & 127;
        sq[i] = base[tok * 384 + d];
        sk[i] = base[tok * 384 + 128 + d];
        sv[i] = base[tok * 384 + 256 + d];
    }
    __syncthreads();
    int hh = tid >> 5, i = tid & 31;
    float q[16];
#pragma unroll
    for (int d = 0; d < 16; ++d) q[d] = sq[i * 128 + hh * 16 + d];
    float s[32];
    float mx = -1e30f;
#pragma unroll
    for (int j = 0; j < 32; ++j) {
        float dot = 0.0f;
#pragma unroll
        for (int d = 0; d < 16; ++d) dot += q[d] * sk[j * 128 + hh * 16 + d];
        dot *= 0.25f; // 1/sqrt(16)
        s[j] = dot;
        mx = fmaxf(mx, dot);
    }
    float sum = 0.0f;
#pragma unroll
    for (int j = 0; j < 32; ++j) { s[j] = __expf(s[j] - mx); sum += s[j]; }
    float inv = 1.0f / sum;
    float acc[16];
#pragma unroll
    for (int d = 0; d < 16; ++d) acc[d] = 0.0f;
#pragma unroll
    for (int j = 0; j < 32; ++j) {
        float p = s[j] * inv;
#pragma unroll
        for (int d = 0; d < 16; ++d) acc[d] += p * sv[j * 128 + hh * 16 + d];
    }
#pragma unroll
    for (int d = 0; d < 16; ++d)
        SA[(size_t)b * KS * DH + (size_t)i * DH + hh * 16 + d] = acc[d];
}

// =====================================================================
// h_global scatter: per graph, OUT[n,:] = HL[n,:] + Wsum[b][n][:] @ vns[b]
// WMMA: M=1024, K=32, N=128;  vns pre-swizzled in LDS
// =====================================================================
__global__ __launch_bounds__(256)
void hglobal_kernel(const float* __restrict__ Wsum, const float* __restrict__ VNS,
                    const float* __restrict__ HL, float* __restrict__ OUT) {
    __shared__ bf16 sV[32 * 128];   // B-swizzled
    const int b = blockIdx.x, tid = threadIdx.x;
    for (int i = tid; i < 32 * 128; i += 256) {
        int k = i >> 7, d = i & 127;
        sV[bswz(k, d, 128)] = (bf16)VNS[(size_t)b * KS * DH + i];
    }
    __syncthreads();
    const int wave = tid >> 5, lane = tid & 31, kh = lane >> 4, l16 = lane & 15;
    const float* Wg = Wsum + (size_t)b * NPER * KS;
    for (int j = 0; j < 8; ++j) {
        int mt = wave * 8 + j;
        const float* ar = Wg + (size_t)(mt * 16 + l16) * KS + kh * 8;
        float4 r0 = *(const float4*)(ar);
        float4 r1 = *(const float4*)(ar + 4);
        float4 r2 = *(const float4*)(ar + 16);
        float4 r3 = *(const float4*)(ar + 20);
        v16bf af;
        af[0] = (bf16)r0.x; af[1] = (bf16)r0.y; af[2]  = (bf16)r0.z; af[3]  = (bf16)r0.w;
        af[4] = (bf16)r1.x; af[5] = (bf16)r1.y; af[6]  = (bf16)r1.z; af[7]  = (bf16)r1.w;
        af[8] = (bf16)r2.x; af[9] = (bf16)r2.y; af[10] = (bf16)r2.z; af[11] = (bf16)r2.w;
        af[12] = (bf16)r3.x; af[13] = (bf16)r3.y; af[14] = (bf16)r3.z; af[15] = (bf16)r3.w;
#pragma unroll
        for (int nt = 0; nt < 8; ++nt) {
            v16bf bv = ld_frag16(&sV[(nt * 32 + lane) * 16]);
            v8f c = zero8();
            c = wmma_bf(af, bv, c);
#pragma unroll
            for (int r = 0; r < 8; ++r) {
                int n = mt * 16 + kh * 8 + r;
                int d = nt * 16 + l16;
                size_t gi = (size_t)b * NPER * DH + (size_t)n * DH + d;
                OUT[gi] = HL[gi] + c[r];
            }
        }
    }
}

// =====================================================================
// Host side
// =====================================================================
template <int KD, int NC>
static void launch_gemm(const float* A, const float* W, int transW,
                        const float* bias, int M, int act,
                        const float* resid, float* C, bf16* Cbf, hipStream_t s) {
    int smem = (KD * NC + 64 * KD) * (int)sizeof(bf16);
    (void)hipFuncSetAttribute((const void*)gemm_kernel<KD, NC>,
                              hipFuncAttributeMaxDynamicSharedMemorySize, smem);
    int blocks = (M + 63) / 64;
    gemm_kernel<KD, NC><<<blocks, 256, smem, s>>>(A, W, transW, bias, M, act, resid, C, Cbf);
}

extern "C" void kernel_launch(void* const* d_in, const int* in_sizes, int n_in,
                              void* d_out, int out_size, void* d_ws, size_t ws_size,
                              hipStream_t stream) {
    (void)in_sizes; (void)n_in; (void)out_size; (void)ws_size;

    const float* x     = (const float*)d_in[0];
    const float* eattr = (const float*)d_in[1];
    const int*   ei    = (const int*)d_in[2];
    const int* src = ei;
    const int* dst = ei + EE;
    // d_in[3] = node_graph_id (unused: graphs are equal-size, contiguous)

    // pma params
    const float* seed  = (const float*)d_in[25];
    const float* Wq    = (const float*)d_in[26];
    const float* bq    = (const float*)d_in[27];
    const float* Wk    = (const float*)d_in[28];
    const float* bk    = (const float*)d_in[29];
    const float* Wv    = (const float*)d_in[30];
    const float* bv    = (const float*)d_in[31];
    const float* Wop   = (const float*)d_in[32];
    const float* bop   = (const float*)d_in[33];
    const float* ln0g  = (const float*)d_in[34];
    const float* ln0b  = (const float*)d_in[35];
    const float* ln1g  = (const float*)d_in[36];
    const float* ln1b  = (const float*)d_in[37];
    // attn params
    const float* Wqkv  = (const float*)d_in[38];
    const float* bqkv  = (const float*)d_in[39];
    const float* AWo   = (const float*)d_in[40];
    const float* Abo   = (const float*)d_in[41];
    // ffn params
    const float* F1    = (const float*)d_in[42];
    const float* fb1   = (const float*)d_in[43];
    const float* F2    = (const float*)d_in[44];
    const float* fb2   = (const float*)d_in[45];

    // ---- workspace layout ----
    float* h      = (float*)d_ws;                       // N*D
    float* aggr   = h    + (size_t)ND * DH;             // N*D
    float* z      = aggr + (size_t)ND * DH;             // N*D
    float* hl     = z    + (size_t)ND * DH;             // N*D
    float* bnsums = hl   + (size_t)ND * DH;             // 256 floats

    // K/V bf16 alias the z region (z is dead after the last BN apply)
    bf16* Kb = (bf16*)z;
    bf16* Vb = Kb + (size_t)ND * DH;

    // small buffers alias the h region (h is dead after layer-2 MLP)
    float* Wsum = h;                                    // B*1024*32
    float* Q    = Wsum + (size_t)BB * NPER * KS;        // 32*128
    bf16*  Qbf  = (bf16*)(Q + KS * DH);                 // 32*128 bf16
    float* sp   = Q + KS * DH + (KS * DH) / 2;
    float* o    = sp;  sp += (size_t)BB * KS * DH;
    float* oln  = sp;  sp += (size_t)BB * KS * DH;
    float* o2   = sp;  sp += (size_t)BB * KS * DH;
    float* vns  = sp;  sp += (size_t)BB * KS * DH;
    float* qkv  = sp;  sp += (size_t)BB * KS * 384;
    float* sa   = sp;  sp += (size_t)BB * KS * DH;
    float* vns2 = sp;  sp += (size_t)BB * KS * DH;
    float* mid  = sp;  sp += (size_t)BB * KS * 256;
    float* vnsf = sp;  sp += (size_t)BB * KS * DH;

    const int GINE_SMEM = (64 * 128 + 128 * 128 + 128 * 128 + 64 * 128) * (int)sizeof(bf16)
                          + 256 * (int)sizeof(float);
    (void)hipFuncSetAttribute((const void*)gine_mlp_kernel,
                              hipFuncAttributeMaxDynamicSharedMemorySize, GINE_SMEM);
    const int PMA_SMEM = 32 * 1024 * (int)sizeof(float) + 2 * 32 * 1024 * (int)sizeof(bf16);
    (void)hipFuncSetAttribute((const void*)pma_attn_kernel,
                              hipFuncAttributeMaxDynamicSharedMemorySize, PMA_SMEM);

    // ---- 3x GINEConv + BN ----
    for (int l = 0; l < 3; ++l) {
        const float* W1    = (const float*)d_in[4 + 7 * l];
        const float* b1    = (const float*)d_in[5 + 7 * l];
        const float* W2    = (const float*)d_in[6 + 7 * l];
        const float* b2    = (const float*)d_in[7 + 7 * l];
        const float* eps   = (const float*)d_in[8 + 7 * l];
        const float* gamma = (const float*)d_in[9 + 7 * l];
        const float* beta  = (const float*)d_in[10 + 7 * l];
        const float* hcur  = (l == 0) ? x : h;

        hipMemsetAsync(aggr, 0, (size_t)ND * DH * sizeof(float), stream);
        edge_kernel<<<EE / 8, 256, 0, stream>>>(hcur, eattr, src, dst, aggr);
        hipMemsetAsync(bnsums, 0, 256 * sizeof(float), stream);
        gine_mlp_kernel<<<ND / 64, 256, GINE_SMEM, stream>>>(hcur, aggr, eps, W1, b1, W2, b2, z, bnsums);
        bn_apply_kernel<<<(ND * DH) / 256, 256, 0, stream>>>(
            z, bnsums, gamma, beta, (l == 2) ? x : nullptr, (l == 2) ? hl : h);
    }

    // ---- PMA pooling ----
    launch_gemm<128, 128>(hl, Wk, 0, bk, ND, 0, nullptr, nullptr, Kb, stream);
    launch_gemm<128, 128>(hl, Wv, 0, bv, ND, 0, nullptr, nullptr, Vb, stream);
    launch_gemm<128, 128>(seed, Wq, 0, bq, KS, 0, nullptr, Q, Qbf, stream);
    pma_attn_kernel<<<BB, 256, PMA_SMEM, stream>>>(Qbf, Q, Kb, Vb, o, Wsum);
    ln_kernel<<<(BB * KS) / 8, 256, 0, stream>>>(o, ln0g, ln0b, oln, BB * KS);
    launch_gemm<128, 128>(oln, Wop, 0, bop, BB * KS, 1, oln, o2, nullptr, stream);
    ln_kernel<<<(BB * KS) / 8, 256, 0, stream>>>(o2, ln1g, ln1b, vns, BB * KS);

    // ---- global transformer on VNs ----
    launch_gemm<128, 384>(vns, Wqkv, 1, bqkv, BB * KS, 0, nullptr, qkv, nullptr, stream);
    attn_small_kernel<<<BB, 256, 0, stream>>>(qkv, sa);
    launch_gemm<128, 128>(sa, AWo, 1, Abo, BB * KS, 0, nullptr, vns2, nullptr, stream);

    // ---- FFN residual ----
    launch_gemm<128, 256>(vns2, F1, 0, fb1, BB * KS, 1, nullptr, mid, nullptr, stream);
    launch_gemm<256, 128>(mid, F2, 0, fb2, BB * KS, 0, vns2, vnsf, nullptr, stream);

    // ---- scatter VN embeddings back + residual ----
    hglobal_kernel<<<BB, 256, 0, stream>>>(Wsum, vnsf, hl, (float*)d_out);
}
